// Net_53034256171343
// MI455X (gfx1250) — matrix operand order repaired and checked by reading, not compiled, not run on hardware
//
#include <hip/hip_runtime.h>
#include <stdint.h>

#define NN 20000        // nodes
#define NE 320000       // edges
#define INF_ 64         // input node feature dim
#define FE0 112         // layer-1 edge feature dim
#define HID 50
#define NREST 9
#define EPS 1e-5f

typedef __attribute__((ext_vector_type(16))) __bf16 v16bf;
typedef __attribute__((ext_vector_type(8)))  float  v8f;
typedef __attribute__((ext_vector_type(4)))  uint32_t v4u;

union Frag { uint32_t u[8]; v16bf v; };

__device__ __forceinline__ uint32_t f32_to_bf16_bits(float f) {
    union { float f; uint32_t u; } c; c.f = f;
    return ((c.u + 0x7FFFu + ((c.u >> 16) & 1u)) >> 16) & 0xFFFFu;
}

// packed f32x2 -> bf16x2 (clang emits v_cvt_pk_bf16_f32)
__device__ __forceinline__ uint32_t pack2bf(float v0, float v1) {
    union { __bf16 h[2]; uint32_t u; } p;
    p.h[0] = (__bf16)v0;
    p.h[1] = (__bf16)v1;
    return p.u;
}

// async global -> LDS copy of 16 bytes (gfx1250 ASYNC path, ASYNCcnt tracked)
__device__ __forceinline__ void async_copy_b128(const v4u* gsrc, v4u* ldst) {
    const uint32_t loff = (uint32_t)(uintptr_t)ldst;   // LDS offset = low 32 bits
    asm volatile("global_load_async_to_lds_b128 %0, %1, off"
                 :: "v"(loff), "v"(gsrc) : "memory");
}
__device__ __forceinline__ void async_wait0() {
    asm volatile("s_wait_asynccnt 0x0" ::: "memory");
}

// ---------------------------------------------------------------------------
// Pack fp32 [dk x 50] weights into bf16 WMMA B-fragments:
// frag index f = ((kt*4 + nt)*32 + lane)*8 + v
// lane: n = nt*16 + (lane&15); K = kt*32 + ((lane>>4)&1)*16 + 2v (+1 in hi half)
// Zero-pad K beyond dk and N beyond 50.
// ---------------------------------------------------------------------------
__global__ __launch_bounds__(256)
void prepack(const float* __restrict__ W1m, const float* __restrict__ W1e,
             const float* __restrict__ Wm,  const float* __restrict__ We,
             uint32_t* __restrict__ packed)
{
    const int id = blockIdx.x * 256 + threadIdx.x;
    const int TOT = 2 * 8192 + 18 * 5120;   // layer1 (Ktiles=8) + 9 layers x2 (Ktiles=5)
    if (id >= TOT) return;

    const float* W; int dk, f;
    if (id < 8192)        { W = W1m; dk = 240; f = id; }
    else if (id < 16384)  { W = W1e; dk = 240; f = id - 8192; }
    else {
        int idr   = id - 16384;
        int layer = idr / 10240;
        int rem   = idr % 10240;
        int which = rem / 5120;
        f  = rem % 5120;
        dk = 150;
        W  = (which == 0 ? Wm : We) + (size_t)layer * 150 * HID;
    }
    const int v    = f & 7;
    const int lane = (f >> 3) & 31;
    const int nt   = (f >> 8) & 3;
    const int kt   = f >> 10;
    const int k    = kt * 32 + ((lane >> 4) & 1) * 16 + 2 * v;
    const int n    = nt * 16 + (lane & 15);
    const float w0 = (k     < dk && n < HID) ? W[(size_t)k       * HID + n] : 0.f;
    const float w1 = (k + 1 < dk && n < HID) ? W[(size_t)(k + 1) * HID + n] : 0.f;
    packed[id] = f32_to_bf16_bits(w0) | (f32_to_bf16_bits(w1) << 16);
}

// ---------------------------------------------------------------------------
// Per-edge dual GEMM: h = [node[dst] | node[src] | ein]  (K = 2F+Fe, padded)
//   msg  = h @ Wm + bm  -> atomicAdd into xacc[dst]   (segment_sum)
//   eout = h @ We + be  -> stored per edge (in-place safe)
// One wave per 16-edge tile; 8 waves / block; NE/128 blocks (exact).
// B fragments async-staged once per block into LDS; A tiles wave-private.
// ---------------------------------------------------------------------------
template <int KT>
__global__ __launch_bounds__(256)
void edge_gemm(const float* __restrict__ nodef, int F,
               const float* __restrict__ ein,   int Fe,
               const int* __restrict__ dst, const int* __restrict__ src,
               const uint32_t* __restrict__ pWm, const uint32_t* __restrict__ pWe,
               const float* __restrict__ bm, const float* __restrict__ be,
               float* __restrict__ xacc, float* __restrict__ eout)
{
    // A staging: 8 waves x 16 rows x 128 dwords (bf16 pairs), XOR-swizzled
    __shared__ uint32_t stage[8][16][128];           // 64 KB
    // B staging: [msg | edge], each KT*4 frags x 32 lanes x 8 dwords
    __shared__ uint32_t bsh[2 * KT * 1024];          // KT*8 KB

    const int lane = threadIdx.x & 31;
    const int wv   = threadIdx.x >> 5;
    const int e0   = blockIdx.x * 128 + wv * 16;
    const int Kcat = 2 * F + Fe;

    // ---- cooperative B copy: global -> LDS via async engine ----
    {
        const v4u* gm = (const v4u*)pWm;
        const v4u* ge = (const v4u*)pWe;
        v4u* lm = (v4u*)bsh;
        v4u* le = (v4u*)(bsh + KT * 1024);
#pragma unroll
        for (int t = 0; t < KT; ++t) {
            const int i = threadIdx.x + t * 256;
            async_copy_b128(gm + i, lm + i);
            async_copy_b128(ge + i, le + i);
        }
    }

    // ---- gather + convert + stage A (dword index d ^ ((row&15)<<2)) ----
    for (int r = 0; r < 16; ++r) {
        const int ei = e0 + r;
        const int dn = dst[ei];
        const int sn = src[ei];
        const int swr = (r & 15) << 2;
        for (int d = lane; d < KT * 16; d += 32) {
            const int k = 2 * d;
            float v0 = 0.f, v1 = 0.f;
            if (k < F)              v0 = nodef[(size_t)dn * F + k];
            else if (k < 2 * F)     v0 = nodef[(size_t)sn * F + (k - F)];
            else if (k < Kcat)      v0 = ein[(size_t)ei * Fe + (k - 2 * F)];
            const int k1 = k + 1;
            if (k1 < F)             v1 = nodef[(size_t)dn * F + k1];
            else if (k1 < 2 * F)    v1 = nodef[(size_t)sn * F + (k1 - F)];
            else if (k1 < Kcat)     v1 = ein[(size_t)ei * Fe + (k1 - 2 * F)];
            stage[wv][r][d ^ swr] = pack2bf(v0, v1);
        }
    }
    async_wait0();          // drain this wave's async B transfers
    __syncthreads();        // make B + all A tiles visible block-wide

    v8f accM[4], accE[4];
    const v8f vzero = {0.f,0.f,0.f,0.f,0.f,0.f,0.f,0.f};
#pragma unroll
    for (int nt = 0; nt < 4; ++nt) { accM[nt] = vzero; accE[nt] = vzero; }

    // A-fragment addressing (this code's contiguous k order, 16-bit A 16x32):
    //  lanes 0-15 : row m=lane,    K {0..15}
    //  lanes 16-31: row m=lane-16, K {16..31}
    // Swizzle only touches dword-index bits [5:2] => 4-dword groups stay
    // contiguous: swizzle the base, load b128.
    const int m    = lane & 15;
    const int koff = (lane & 16) ? 8 : 0;        // dword offset of this lane half's 16 K
    const int sw   = m << 2;                     // staging swizzle for this row
    const uint32_t* rowp = &stage[wv][m][0];
    const uint32_t* bshM = bsh;
    const uint32_t* bshE = bsh + KT * 1024;

#pragma unroll
    for (int kt = 0; kt < KT; ++kt) {
        Frag a;
        *(v4u*)&a.u[0] = *(const v4u*)&rowp[(kt * 16 +     koff) ^ sw];
        *(v4u*)&a.u[4] = *(const v4u*)&rowp[(kt * 16 + 4 + koff) ^ sw];
#pragma unroll
        for (int nt = 0; nt < 4; ++nt) {
            Frag b;
            const v4u* bp = (const v4u*)(bshM + (((kt * 4 + nt) * 32 + lane) * 8));
            *(v4u*)&b.u[0] = bp[0];
            *(v4u*)&b.u[4] = bp[1];
            accM[nt] = __builtin_amdgcn_wmma_f32_16x16x32_bf16(
                false, a.v, false, b.v, (short)0, accM[nt], false, false);
        }
#pragma unroll
        for (int nt = 0; nt < 4; ++nt) {
            Frag b;
            const v4u* bp = (const v4u*)(bshE + (((kt * 4 + nt) * 32 + lane) * 8));
            *(v4u*)&b.u[0] = bp[0];
            *(v4u*)&b.u[4] = bp[1];
            accE[nt] = __builtin_amdgcn_wmma_f32_16x16x32_bf16(
                false, a.v, false, b.v, (short)0, accE[nt], false, false);
        }
    }

    // ---- D layout: VGPR j -> row (j or j+8), col = (lane&15) + 16*nt ----
    const int rbase = (lane & 16) ? 8 : 0;
    int dstv[8];
#pragma unroll
    for (int j = 0; j < 8; ++j) dstv[j] = dst[e0 + rbase + j];

#pragma unroll
    for (int nt = 0; nt < 4; ++nt) {
        const int c = nt * 16 + (lane & 15);
        if (c < HID) {
            const float bmc = bm[c];
            const float bec = be[c];
#pragma unroll
            for (int j = 0; j < 8; ++j) {
                const int ei = e0 + rbase + j;
                atomicAdd(&xacc[(size_t)dstv[j] * HID + c], accM[nt][j] + bmc);
                eout[(size_t)ei * HID + c] = accE[nt][j] + bec;
            }
        }
    }
}

// ---------------------------------------------------------------------------
// BatchNorm batch statistics: one block per channel.
// ---------------------------------------------------------------------------
__global__ __launch_bounds__(256)
void bn_stats(const float* __restrict__ xacc, float* __restrict__ mu,
              float* __restrict__ rs)
{
    __shared__ float s1[256], s2[256];
    const int c = blockIdx.x;
    float sum = 0.f, sq = 0.f;
    for (int n = threadIdx.x; n < NN; n += 256) {
        const float v = xacc[(size_t)n * HID + c];
        sum += v; sq += v * v;
    }
    s1[threadIdx.x] = sum; s2[threadIdx.x] = sq;
    __syncthreads();
    for (int o = 128; o > 0; o >>= 1) {
        if (threadIdx.x < o) {
            s1[threadIdx.x] += s1[threadIdx.x + o];
            s2[threadIdx.x] += s2[threadIdx.x + o];
        }
        __syncthreads();
    }
    if (threadIdx.x == 0) {
        const float mean = s1[0] * (1.0f / NN);
        const float var  = s2[0] * (1.0f / NN) - mean * mean;
        mu[c] = mean;
        rs[c] = rsqrtf(var + EPS);
    }
}

// ---------------------------------------------------------------------------
// cur = (first ? 0 : cur) + elu(bn(xacc))
// ---------------------------------------------------------------------------
__global__ __launch_bounds__(256)
void bn_apply_elu(const float* __restrict__ xacc, const float* __restrict__ mu,
                  const float* __restrict__ rs, const float* __restrict__ g,
                  const float* __restrict__ bt, float* __restrict__ cur, int first)
{
    const int i = blockIdx.x * 256 + threadIdx.x;
    if (i >= NN * HID) return;
    const int c = i % HID;
    float t = (xacc[i] - mu[c]) * rs[c] * g[c] + bt[c];
    t = (t > 0.f) ? t : expm1f(t);
    if (first) cur[i] = t; else cur[i] += t;
}

// ---------------------------------------------------------------------------
extern "C" void kernel_launch(void* const* d_in, const int* in_sizes, int n_in,
                              void* d_out, int out_size, void* d_ws, size_t ws_size,
                              hipStream_t stream)
{
    const float* x    = (const float*)d_in[0];
    const int*   eix  = (const int*)  d_in[1];
    const float* ea   = (const float*)d_in[2];
    /* d_in[3] distance: unused by reference */
    const float* W1m  = (const float*)d_in[4];
    const float* b1m  = (const float*)d_in[5];
    const float* W1e  = (const float*)d_in[6];
    const float* b1e  = (const float*)d_in[7];
    const float* g1   = (const float*)d_in[8];
    const float* bt1  = (const float*)d_in[9];
    const float* Wm   = (const float*)d_in[10];
    const float* bm   = (const float*)d_in[11];
    const float* We   = (const float*)d_in[12];
    const float* be   = (const float*)d_in[13];
    const float* gam  = (const float*)d_in[14];
    const float* bet  = (const float*)d_in[15];

    const int* src = eix;         // edge_index[0]
    const int* dst = eix + NE;    // edge_index[1]

    char* ws = (char*)d_ws;
    float* ebuf = (float*)ws;  ws += (size_t)NE * HID * sizeof(float);
    float* xacc = (float*)ws;  ws += (size_t)NN * HID * sizeof(float);
    float* mu   = (float*)ws;  ws += 64 * sizeof(float);
    float* rs   = (float*)ws;  ws += 64 * sizeof(float);
    uint32_t* packed = (uint32_t*)ws;

    float* cur = (float*)d_out;   // residual accumulator lives in the output

    const int TOTPK = 2 * 8192 + 18 * 5120;
    prepack<<<(TOTPK + 255) / 256, 256, 0, stream>>>(W1m, W1e, Wm, We, packed);

    // -------- layer 1: K = 2*64 + 112 = 240 -> 8 k-tiles --------
    hipMemsetAsync(xacc, 0, (size_t)NN * HID * sizeof(float), stream);
    edge_gemm<8><<<NE / 128, 256, 0, stream>>>(x, INF_, ea, FE0, dst, src,
                                               packed, packed + 8192,
                                               b1m, b1e, xacc, ebuf);
    bn_stats<<<HID, 256, 0, stream>>>(xacc, mu, rs);
    bn_apply_elu<<<(NN * HID + 255) / 256, 256, 0, stream>>>(xacc, mu, rs, g1, bt1, cur, 1);

    // -------- layers 2..10: K = 3*50 = 150 -> 5 k-tiles --------
    for (int i = 0; i < NREST; ++i) {
        uint32_t* pm = packed + 16384 + (size_t)i * 10240;
        uint32_t* pe = pm + 5120;
        hipMemsetAsync(xacc, 0, (size_t)NN * HID * sizeof(float), stream);
        edge_gemm<5><<<NE / 128, 256, 0, stream>>>(cur, HID, ebuf, HID, dst, src,
                                                   pm, pe, bm + i * HID, be + i * HID,
                                                   xacc, ebuf);
        bn_stats<<<HID, 256, 0, stream>>>(xacc, mu, rs);
        bn_apply_elu<<<(NN * HID + 255) / 256, 256, 0, stream>>>(
            xacc, mu, rs, gam + i * HID, bet + i * HID, cur, 0);
    }
}
